// ConvLayer_1443109011969
// MI455X (gfx1250) — matrix operand order repaired and checked by reading, not compiled, not run on hardware
//
#include <hip/hip_runtime.h>

// Problem sizes (fixed by the reference)
#define NN 8000
#define EE 40000
#define HDIM 64
#define EDIM 16
#define PCOLS 4096           // HDIM*HDIM
#define MTILES (NN / 16)     // 500
#define NTILES (PCOLS / 16)  // 256

typedef __attribute__((ext_vector_type(16))) __bf16 v16bf;
typedef __attribute__((ext_vector_type(8)))  float  v8f;

__device__ __forceinline__ unsigned short f2bf(float f) {
  union { float f; unsigned int u; } v; v.f = f;
  unsigned int r = v.u + 0x7FFFu + ((v.u >> 16) & 1u);  // round-to-nearest-even
  return (unsigned short)(r >> 16);
}
__device__ __forceinline__ float bf2f(unsigned short b) {
  union { unsigned int u; float f; } v; v.u = ((unsigned int)b) << 16;
  return v.f;
}

// Pack one fp32 value at logical (row n, k=o) into the wave32 bf16 A-fragment
// layout for V_WMMA_F32_16X16X32_BF16 (CDNA5 ISA "16-bit A-Matrix 16x32").
__device__ __forceinline__ void pack_a(unsigned short* Apack, int n, int o, float v) {
  const int kb   = o >> 5;          // which K-block of 32 (K total = 64)
  const int kk   = o & 31;
  const int lane = (n & 15) + ((kk & 8) ? 16 : 0);
  const int elem = (kk & 7) + ((kk >= 16) ? 8 : 0);
  Apack[((((n >> 4) * 2) + kb) * 32 + lane) * 16 + elem] = f2bf(v);
}

__global__ void k_zero(float* __restrict__ p, int n) {
  int i = blockIdx.x * 256 + threadIdx.x;
  if (i < n) p[i] = 0.0f;
}

__global__ void k_count(const int* __restrict__ ei, float* __restrict__ counts) {
  int e = blockIdx.x * 256 + threadIdx.x;
  if (e < EE) atomicAdd(&counts[ei[EE + e]], 1.0f);
}

// state(=d_out) <- x ; Apack <- bf16 fragments of x
__global__ void k_init(const float* __restrict__ x, float* __restrict__ state,
                       unsigned short* __restrict__ Apack) {
  int idx = blockIdx.x * 256 + threadIdx.x;      // exactly NN*HDIM threads
  int n = idx >> 6, o = idx & 63;
  float v = x[idx];
  state[idx] = v;
  pack_a(Apack, n, o, v);
}

// Pack W2 (permuted: contraction index i becomes GEMM-K) into bf16 B-fragments.
// B element (k, c) with c = h*64+o reads W2[h*4096 + k*64 + o].
__global__ void k_w2pack(const float* __restrict__ W2, unsigned short* __restrict__ Bpack) {
  int idx = blockIdx.x * 256 + threadIdx.x;      // 64*4096 = 262144 threads
  int elem  = idx & 15;
  int lane  = (idx >> 4) & 31;
  int kb    = (idx >> 9) & 1;
  int ntile = idx >> 10;
  int ncol  = lane & 15;
  int kk    = ((lane >> 4) << 3) + (elem & 7) + ((elem >> 3) << 4);  // lanes>=16 hold K 8..15, 24..31
  int k     = kb * 32 + kk;                      // i index (GEMM K), 0..63
  int c     = ntile * 16 + ncol;                 // 0..4095
  int h     = c >> 6, o = c & 63;
  Bpack[idx] = f2bf(W2[h * PCOLS + k * HDIM + o]);
}

// hrelu = relu(edge_attr @ W1 + b1)   [E,64], loop-invariant
__global__ void k_hrelu(const float* __restrict__ ea, const float* __restrict__ W1,
                        const float* __restrict__ b1, float* __restrict__ hrelu) {
  int idx = blockIdx.x * 256 + threadIdx.x;      // EE*HDIM threads exactly
  int e = idx >> 6, o = idx & 63;
  const float* a = ea + e * EDIM;
  float acc = b1[o];
#pragma unroll
  for (int i = 0; i < EDIM; ++i) acc = fmaf(a[i], W1[i * HDIM + o], acc);
  hrelu[idx] = fmaxf(acc, 0.0f);
}

// P(bf16) = state(bf16) @ W2' via v_wmma_f32_16x16x32_bf16.
// One wave computes a 16x64 strip: A fragments reused across 4 N-tiles.
__global__ void k_gemm(const unsigned short* __restrict__ Apack,
                       const unsigned short* __restrict__ Bpack,
                       unsigned short* __restrict__ Pb) {
  const int wave = blockIdx.x * 8 + (threadIdx.x >> 5);  // 32000 waves total
  const int lane = threadIdx.x & 31;
  const int mtile = wave >> 6;                           // 0..499
  const int g     = wave & 63;                           // group of 4 ntiles
  if (mtile >= MTILES) return;                           // wave-uniform

  const v16bf a0 = *(const v16bf*)(Apack + (size_t)(((mtile * 2 + 0) * 32) + lane) * 16);
  const v16bf a1 = *(const v16bf*)(Apack + (size_t)(((mtile * 2 + 1) * 32) + lane) * 16);

  const int ncol  = lane & 15;
  const int mbase = mtile * 16 + ((lane >> 4) << 3);     // D rows: j (+8 for hi lanes)

#pragma unroll
  for (int t = 0; t < 4; ++t) {
    const int ntile = g * 4 + t;
    const v16bf b0 = *(const v16bf*)(Bpack + (size_t)(((ntile * 2 + 0) * 32) + lane) * 16);
    const v16bf b1 = *(const v16bf*)(Bpack + (size_t)(((ntile * 2 + 1) * 32) + lane) * 16);
    v8f c = {0.f, 0.f, 0.f, 0.f, 0.f, 0.f, 0.f, 0.f};
    c = __builtin_amdgcn_wmma_f32_16x16x32_bf16(false, a0, false, b0, (short)0, c, false, false);
    c = __builtin_amdgcn_wmma_f32_16x16x32_bf16(false, a1, false, b1, (short)0, c, false, false);
    unsigned short* outp = Pb + (size_t)mbase * PCOLS + ntile * 16 + ncol;
#pragma unroll
    for (int j = 0; j < 8; ++j) outp[(size_t)j * PCOLS] = f2bf(c[j]);
  }
}

// Per-edge message: msg[o] = sum_i h[src,i]*b2[i*64+o] + sum_h hrelu[e,h]*P[src,h*64+o]
// scatter-added into agg[dst]. P is bf16 and L2-resident (65.5 MB < 192 MB L2).
__global__ void k_edge(const int* __restrict__ ei, const float* __restrict__ hrelu,
                       const unsigned short* __restrict__ Pb, const float* __restrict__ state,
                       const float* __restrict__ b2, float* __restrict__ agg) {
  const int e = blockIdx.x * 4 + (threadIdx.x >> 6);     // EE/4 blocks exactly
  const int o = threadIdx.x & 63;
  const int s = ei[e];
  const int d = ei[EE + e];
  const unsigned short* Ps = Pb + (size_t)s * PCOLS;
  __builtin_prefetch(Ps, 0, 0);                          // global_prefetch_b8
  const float* hs = state + s * HDIM;
  const float* hr = hrelu + (size_t)e * HDIM;
  float acc = 0.0f;
#pragma unroll 8
  for (int i = 0; i < HDIM; ++i) acc = fmaf(hs[i], b2[i * HDIM + o], acc);
#pragma unroll 8
  for (int h = 0; h < HDIM; ++h) acc = fmaf(hr[h], bf2f(Ps[h * HDIM + o]), acc);
  atomicAdd(agg + (size_t)d * HDIM + o, acc);
}

// Fused node update: m = relu(agg/cnt + h@W_root^T + b_conv); GRU cell; state<-h';
// also repack h' into bf16 A-fragments for the next iteration's WMMA GEMM.
__global__ void k_node(float* __restrict__ state, const float* __restrict__ agg,
                       const float* __restrict__ counts, const float* __restrict__ W_root,
                       const float* __restrict__ b_conv, const float* __restrict__ w_ih,
                       const float* __restrict__ w_hh, const float* __restrict__ b_ih,
                       const float* __restrict__ b_hh, unsigned short* __restrict__ Apack) {
  __shared__ float sh_h[4 * HDIM];
  __shared__ float sh_m[4 * HDIM];
  const int l = threadIdx.x >> 6;                        // local node 0..3
  const int o = threadIdx.x & 63;
  const int n = blockIdx.x * 4 + l;                      // NN/4 blocks exactly

  const float hval = state[n * HDIM + o];
  sh_h[l * HDIM + o] = hval;
  __syncthreads();
  const float* hv = sh_h + l * HDIM;

  const float inv = 1.0f / fmaxf(counts[n], 1.0f);
  float acc = b_conv[o] + agg[n * HDIM + o] * inv;
#pragma unroll 8
  for (int i = 0; i < HDIM; ++i) acc = fmaf(hv[i], W_root[o * HDIM + i], acc);
  const float m = fmaxf(acc, 0.0f);
  sh_m[l * HDIM + o] = m;
  __syncthreads();
  const float* mv = sh_m + l * HDIM;

  float gir = b_ih[o], giz = b_ih[HDIM + o], gin = b_ih[2 * HDIM + o];
  float ghr = b_hh[o], ghz = b_hh[HDIM + o], ghn = b_hh[2 * HDIM + o];
#pragma unroll 4
  for (int i = 0; i < HDIM; ++i) {
    const float mi = mv[i], hi = hv[i];
    gir = fmaf(mi, w_ih[o * HDIM + i], gir);
    giz = fmaf(mi, w_ih[(HDIM + o) * HDIM + i], giz);
    gin = fmaf(mi, w_ih[(2 * HDIM + o) * HDIM + i], gin);
    ghr = fmaf(hi, w_hh[o * HDIM + i], ghr);
    ghz = fmaf(hi, w_hh[(HDIM + o) * HDIM + i], ghz);
    ghn = fmaf(hi, w_hh[(2 * HDIM + o) * HDIM + i], ghn);
  }
  const float r = 1.0f / (1.0f + __expf(-(gir + ghr)));
  const float z = 1.0f / (1.0f + __expf(-(giz + ghz)));
  const float nn_ = tanhf(gin + r * ghn);
  const float hnew = (1.0f - z) * nn_ + z * hval;

  state[n * HDIM + o] = hnew;
  pack_a(Apack, n, o, hnew);
}

extern "C" void kernel_launch(void* const* d_in, const int* in_sizes, int n_in,
                              void* d_out, int out_size, void* d_ws, size_t ws_size,
                              hipStream_t stream) {
  const float* x         = (const float*)d_in[0];
  const float* edge_attr = (const float*)d_in[1];
  const float* W1        = (const float*)d_in[2];
  const float* b1        = (const float*)d_in[3];
  const float* W2        = (const float*)d_in[4];
  const float* b2        = (const float*)d_in[5];
  const float* W_root    = (const float*)d_in[6];
  const float* b_conv    = (const float*)d_in[7];
  const float* w_ih      = (const float*)d_in[8];
  const float* w_hh      = (const float*)d_in[9];
  const float* b_ih      = (const float*)d_in[10];
  const float* b_hh      = (const float*)d_in[11];
  const int*   ei        = (const int*)d_in[12];
  float* state = (float*)d_out;                          // out == hidden in reference

  // Workspace carve-out (256B aligned); ~79 MB total.
  char* ws = (char*)d_ws;
  auto take = [&](size_t bytes) { char* p = ws; ws += (bytes + 255) & ~(size_t)255; return p; };
  float*          counts = (float*)take((size_t)NN * 4);
  float*          agg    = (float*)take((size_t)NN * HDIM * 4);
  float*          hrelu  = (float*)take((size_t)EE * HDIM * 4);
  unsigned short* Apack  = (unsigned short*)take((size_t)NN * HDIM * 2);
  unsigned short* Bpack  = (unsigned short*)take((size_t)HDIM * PCOLS * 2);
  unsigned short* Pb     = (unsigned short*)take((size_t)NN * PCOLS * 2);

  // Loop-invariant setup
  k_zero  <<<(NN + 255) / 256, 256, 0, stream>>>(counts, NN);
  k_count <<<(EE + 255) / 256, 256, 0, stream>>>(ei, counts);
  k_init  <<<(NN * HDIM) / 256, 256, 0, stream>>>(x, state, Apack);
  k_w2pack<<<(HDIM * PCOLS) / 256, 256, 0, stream>>>(W2, Bpack);
  k_hrelu <<<(EE * HDIM) / 256, 256, 0, stream>>>(edge_attr, W1, b1, hrelu);

  for (int it = 0; it < 3; ++it) {
    k_zero<<<(NN * HDIM + 255) / 256, 256, 0, stream>>>(agg, NN * HDIM);
    k_gemm<<<(MTILES * 64) / 8, 256, 0, stream>>>(Apack, Bpack, Pb);      // 32000 waves
    k_edge<<<EE / 4, 256, 0, stream>>>(ei, hrelu, Pb, state, b2, agg);
    k_node<<<NN / 4, 256, 0, stream>>>(state, agg, counts, W_root, b_conv,
                                       w_ih, w_hh, b_ih, b_hh, Apack);
  }
}
